// Attention_24283745092396
// MI455X (gfx1250) — matrix operand order repaired and checked by reading, not compiled, not run on hardware
//
#include <hip/hip_runtime.h>
#include <hip/hip_bf16.h>
#include <math.h>

typedef __bf16 bf16;
typedef __attribute__((ext_vector_type(16))) __bf16 v16bf;
typedef __attribute__((ext_vector_type(8)))  __bf16 v8bf;
typedef __attribute__((ext_vector_type(4)))  __bf16 v4bf;
typedef __attribute__((ext_vector_type(8)))  float  v8f;
typedef __attribute__((ext_vector_type(4)))  float  v4f;

static constexpr int B = 32, T = 2048, D = 1024, H = 1024;
static constexpr int TT = 64;                 // t rows per block in energy kernel
static constexpr int LDS_STRIDE = D + 8;      // bf16 elems; +16B pad -> conflict-free b128 LDS reads

// gfx1250 has V_TANH_F32 (1 transcendental op). Fall back to a branch-free
// exp-based form if the builtin is missing.
__device__ __forceinline__ float fast_tanh(float x) {
#if __has_builtin(__builtin_amdgcn_tanhf)
    return __builtin_amdgcn_tanhf(x);
#elif __has_builtin(__builtin_amdgcn_tanh_f32)
    return __builtin_amdgcn_tanh_f32(x);
#else
    const float e = __expf(-2.0f * __builtin_fabsf(x));
    const float t = __builtin_fmaf(-2.0f, e / (1.0f + e), 1.0f);
    return __builtin_copysignf(t, x);
#endif
}

// concat two 8-wide bf16 vectors into a 16-wide WMMA operand
__device__ __forceinline__ v16bf cat16(v8bf lo, v8bf hi) {
    return __builtin_shufflevector(lo, hi, 0,1,2,3,4,5,6,7,8,9,10,11,12,13,14,15);
}

// ---------------------------------------------------------------------------
// Kernel 1: Ut[n][d] = bf16(U[d][n])   (transpose+convert so WMMA B-fragments
// become contiguous K-runs at fixed N)
// ---------------------------------------------------------------------------
__global__ __launch_bounds__(256) void transpose_u_kernel(
    const float* __restrict__ U, bf16* __restrict__ Ut) {
    __shared__ float tile[32][33];
    const int n0 = blockIdx.x * 32;
    const int d0 = blockIdx.y * 32;
    const int tx = threadIdx.x & 31;
    const int ty = threadIdx.x >> 5;   // 0..7
#pragma unroll
    for (int i = 0; i < 32; i += 8)
        tile[ty + i][tx] = U[(size_t)(d0 + ty + i) * H + (n0 + tx)];
    __syncthreads();
#pragma unroll
    for (int i = 0; i < 32; i += 8)
        Ut[(size_t)(n0 + ty + i) * D + (d0 + tx)] = (bf16)tile[tx][ty + i];
}

// ---------------------------------------------------------------------------
// Kernel 2: ps[b][n] = sum_d s[b][d] * W[d][n]   (tiny: 33M FMA)
// ---------------------------------------------------------------------------
__global__ __launch_bounds__(256) void proj_s_kernel(
    const float* __restrict__ s, const float* __restrict__ W,
    float* __restrict__ ps) {
    const int b = blockIdx.x;
    const int n = blockIdx.y * 256 + threadIdx.x;
    const float* sv = s + (size_t)b * D;
    float acc = 0.0f;
    for (int d = 0; d < D; ++d)
        acc = fmaf(sv[d], W[(size_t)d * H + n], acc);
    ps[b * H + n] = acc;
}

// ---------------------------------------------------------------------------
// Kernel 3 (main): fused  energy[b][t] = tanh(ps[b] + h[b][t]@U) . v
// One block per (64-row t-tile, batch). 8 waves; each wave owns 8 N-tiles
// processed in PAIRS across all 4 M-tiles: per K-step we load 4 A-fragments
// (LDS) + 2 B-fragments (L2) and issue 8 independent WMMAs.
//   -> 1 ds_load_b128 per WMMA (LDS-BW limited otherwise)
//   -> 0.5 global b128 per WMMA
//   -> 8 independent accumulator chains hide WMMA latency / hazard NOPs.
// proj_h never touches memory; tanh + v-dot fused on the accumulator tile.
// ---------------------------------------------------------------------------
__global__ __launch_bounds__(256) void energy_kernel(
    const float* __restrict__ h, const int* __restrict__ mask,
    const bf16* __restrict__ Ut, const float* __restrict__ ps,
    const float* __restrict__ v, float* __restrict__ energy) {
    extern __shared__ char smem_raw[];
    bf16* ldsh = (bf16*)smem_raw;                 // [TT][LDS_STRIDE] bf16
    __shared__ float e_s[TT];
    __shared__ float ps_s[H];
    __shared__ float v_s[H];

    const int ttile = blockIdx.x;
    const int b     = blockIdx.y;
    const int t0    = ttile * TT;
    const int tid   = threadIdx.x;

    if (tid < TT) e_s[tid] = 0.0f;

    // stage ps row + v into LDS (8 KB) so the epilogue uses ds loads
#pragma unroll
    for (int i = tid; i < H; i += 256) {
        ps_s[i] = ps[b * H + i];
        v_s[i]  = v[i];
    }

    // --- stage masked h tile (64 x 1024 f32 -> bf16 LDS), 16B global loads ---
#pragma unroll 4
    for (int i = 0; i < 64; ++i) {                 // 64*256 float4 total
        const int idx = tid + i * 256;
        const int row = idx >> 8;                  // 256 float4 per row
        const int c4  = (idx & 255) << 2;          // starting float col
        const float msk = mask[b * T + t0 + row] ? 1.0f : 0.0f;
        v4f hv = *(const v4f*)(h + ((size_t)b * T + t0 + row) * D + c4);
        v4bf hb;
        hb[0] = (bf16)(hv[0] * msk);
        hb[1] = (bf16)(hv[1] * msk);
        hb[2] = (bf16)(hv[2] * msk);
        hb[3] = (bf16)(hv[3] * msk);
        *(v4bf*)(ldsh + row * LDS_STRIDE + c4) = hb;
    }
    __syncthreads();

    // --- WMMA compute ---
    const int wave  = tid >> 5;
    const int lane  = tid & 31;
    const int hlf   = lane >> 4;      // lane half (K-half selector)
    const int col   = lane & 15;      // A: M row sel / B: N col sel / C/D: N
    const int nbase = wave * 8;       // 8 N-tiles per wave, as 4 pairs

    // A lane base: row M = m*16 + col ; contiguous K run of 16 at +16*hlf
    const bf16* Ap = ldsh + col * LDS_STRIDE + 16 * hlf;

    float eacc[4][8];
#pragma unroll
    for (int m = 0; m < 4; ++m)
#pragma unroll
        for (int r = 0; r < 8; ++r) eacc[m][r] = 0.0f;

#pragma unroll 1
    for (int np = 0; np < 4; ++np) {
        const int ncol0 = (nbase + np * 2) * 16 + col;
        const int ncol1 = ncol0 + 16;
        // B fragment source: column N ; contiguous K run of 16 at +16*hlf
        const bf16* Bp0 = Ut + (size_t)ncol0 * D + 16 * hlf;
        const bf16* Bp1 = Ut + (size_t)ncol1 * D + 16 * hlf;

        v8f c0[4], c1[4];
#pragma unroll
        for (int m = 0; m < 4; ++m) { c0[m] = (v8f){}; c1[m] = (v8f){}; }

#pragma unroll 1
        for (int k0 = 0; k0 < D; k0 += 32) {
            const v16bf bv0 = cat16(*(const v8bf*)(Bp0 + k0), *(const v8bf*)(Bp0 + k0 + 8));
            const v16bf bv1 = cat16(*(const v8bf*)(Bp1 + k0), *(const v8bf*)(Bp1 + k0 + 8));
#pragma unroll
            for (int m = 0; m < 4; ++m) {
                const bf16* a = Ap + (m * 16) * LDS_STRIDE + k0;
                const v16bf av = cat16(*(const v8bf*)(a), *(const v8bf*)(a + 8));
                c0[m] = __builtin_amdgcn_wmma_f32_16x16x32_bf16(
                            false, av, false, bv0, (short)0, c0[m], false, false);
                c1[m] = __builtin_amdgcn_wmma_f32_16x16x32_bf16(
                            false, av, false, bv1, (short)0, c1[m], false, false);
            }
        }
        // fuse: + proj_s, hardware tanh, dot with v
        // (C/D tile: VGPR r -> row r + 8*hlf, col = lane&15)
        const float psv0 = ps_s[ncol0], vv0 = v_s[ncol0];
        const float psv1 = ps_s[ncol1], vv1 = v_s[ncol1];
#pragma unroll
        for (int m = 0; m < 4; ++m)
#pragma unroll
            for (int r = 0; r < 8; ++r) {
                eacc[m][r] += fast_tanh(c0[m][r] + psv0) * vv0;
                eacc[m][r] += fast_tanh(c1[m][r] + psv1) * vv1;
            }
    }

    // reduce partial energies across lanes/waves via LDS float atomics
#pragma unroll
    for (int m = 0; m < 4; ++m)
#pragma unroll
        for (int r = 0; r < 8; ++r)
            atomicAdd(&e_s[m * 16 + 8 * hlf + r], eacc[m][r]);
    __syncthreads();

    if (tid < TT) {
        const int t = t0 + tid;
        energy[(size_t)b * T + t] = mask[b * T + t] ? e_s[tid] : -1e9f;
    }
}

// ---------------------------------------------------------------------------
// Kernel 4: softmax over T per batch
// ---------------------------------------------------------------------------
__global__ __launch_bounds__(256) void softmax_kernel(
    const float* __restrict__ energy, float* __restrict__ score) {
    __shared__ float red[256];
    const int b = blockIdx.x, tid = threadIdx.x;
    const float* e = energy + (size_t)b * T;

    float m = -3.4e38f;
    for (int t = tid; t < T; t += 256) m = fmaxf(m, e[t]);
    red[tid] = m; __syncthreads();
    for (int s = 128; s > 0; s >>= 1) {
        if (tid < s) red[tid] = fmaxf(red[tid], red[tid + s]);
        __syncthreads();
    }
    const float mx = red[0];
    __syncthreads();

    float sum = 0.0f;
    for (int t = tid; t < T; t += 256) sum += __expf(e[t] - mx);
    red[tid] = sum; __syncthreads();
    for (int s = 128; s > 0; s >>= 1) {
        if (tid < s) red[tid] += red[tid + s];
        __syncthreads();
    }
    const float inv = 1.0f / red[0];
    for (int t = tid; t < T; t += 256)
        score[(size_t)b * T + t] = __expf(e[t] - mx) * inv;
}

// ---------------------------------------------------------------------------
// Kernel 5: context[b][d] = sum_t score[b][t] * (mask ? h[b][t][d] : 0)
// HBM-bound second pass over h, fully coalesced.
// ---------------------------------------------------------------------------
__global__ __launch_bounds__(256) void context_kernel(
    const float* __restrict__ h, const int* __restrict__ mask,
    const float* __restrict__ score, float* __restrict__ out) {
    const int b = blockIdx.y;
    const int d = blockIdx.x * 256 + threadIdx.x;
    const float* hb = h + ((size_t)b * T) * D + d;
    float acc = 0.0f;
    for (int t = 0; t < T; ++t) {
        const float sc = score[(size_t)b * T + t];
        const float hv = hb[(size_t)t * D];
        acc += mask[b * T + t] ? sc * hv : 0.0f;
    }
    out[b * D + d] = acc;
}

// ---------------------------------------------------------------------------
extern "C" void kernel_launch(void* const* d_in, const int* in_sizes, int n_in,
                              void* d_out, int out_size, void* d_ws, size_t ws_size,
                              hipStream_t stream) {
    const float* s    = (const float*)d_in[0];
    const float* h    = (const float*)d_in[1];
    const int*   mask = (const int*)  d_in[2];
    const float* W    = (const float*)d_in[3];
    const float* U    = (const float*)d_in[4];
    const float* v    = (const float*)d_in[5];
    float* out = (float*)d_out;

    char* ws = (char*)d_ws;
    bf16*  Ut     = (bf16*) (ws);                                       // 2 MB
    float* ps     = (float*)(ws + (size_t)2*1024*1024);                 // 128 KB
    float* energy = (float*)(ws + (size_t)2*1024*1024 + 128*1024);      // 256 KB
    float* score  = (float*)(ws + (size_t)2*1024*1024 + 384*1024);      // 256 KB

    hipLaunchKernelGGL(transpose_u_kernel, dim3(H/32, D/32), dim3(256), 0, stream, U, Ut);
    hipLaunchKernelGGL(proj_s_kernel,      dim3(B, H/256),   dim3(256), 0, stream, s, W, ps);

    const size_t smem = (size_t)TT * LDS_STRIDE * sizeof(bf16);         // ~129 KB dynamic LDS
    hipLaunchKernelGGL(energy_kernel, dim3(T/TT, B), dim3(256), smem, stream,
                       h, mask, Ut, ps, v, energy);

    hipLaunchKernelGGL(softmax_kernel, dim3(B),        dim3(256), 0, stream, energy, score);
    hipLaunchKernelGGL(context_kernel, dim3(D/256, B), dim3(256), 0, stream, h, mask, score, out);
}
